// GNNLayer_43370579755265
// MI455X (gfx1250) — matrix operand order repaired and checked by reading, not compiled, not run on hardware
//
#include <hip/hip_runtime.h>

#define Dk 128          // feature dim per segment
#define KTOT 384        // 3 * Dk concat width for every GEMM
#define APAD 392        // padded LDS row stride in halfs (392*2B = 784B -> bank-conflict-free)
#define BLOCKT 256      // 8 waves of 32
#define NKS (KTOT / 32) // 12 WMMA K-steps

typedef __attribute__((ext_vector_type(16))) _Float16 v16h;
typedef __attribute__((ext_vector_type(8)))  _Float16 v8h;
typedef __attribute__((ext_vector_type(8)))  float    v8f;

union V16U { v16h v; v8h h[2]; };

// A-fragment (16x32 f16, ISA layout): lane holds K {base..base+7} and {base+16..base+23}
__device__ __forceinline__ v16h load_a16(const _Float16* p) {
  V16U u;
  u.h[0] = *(const v8h*)(p);
  u.h[1] = *(const v8h*)(p + 16);
  return u.v;
}
// B-fragment (32x16 f16): lane holds K {base..base+7} and {base+16..base+23} for its output column
__device__ __forceinline__ v16h load_b16(const _Float16* p) {
  V16U u;
  u.h[0] = *(const v8h*)(p);
  u.h[1] = *(const v8h*)(p + 16);
  return u.v;
}

// 16x16 f32 tile = A(16x384 in LDS, padded rows) x B(384x16 f16-transposed weights).
// A fragments are prefetched GROUP at a time into distinct registers so many
// ds_load_b128 are in flight per s_wait; B loads are tile-loop invariant and are
// hoisted into registers by the compiler.
template <int GROUP>
__device__ __forceinline__ v8f gemm_tile(const _Float16* Arow, const _Float16* Bp, int aoff) {
  v8f acc = {};
#pragma unroll
  for (int g = 0; g < NKS / GROUP; ++g) {
    v16h a[GROUP];
#pragma unroll
    for (int i = 0; i < GROUP; ++i)
      a[i] = load_a16(Arow + (g * GROUP + i) * 32 + aoff);
#pragma unroll
    for (int i = 0; i < GROUP; ++i) {
      v16h bv = load_b16(Bp + (g * GROUP + i) * 32);
      acc = __builtin_amdgcn_wmma_f32_16x16x32_f16(false, a[i], false, bv, (short)0, acc,
                                                   false, false);
    }
  }
  return acc;
}

__device__ __forceinline__ void atomAddF(float* p, float v) {
  unsafeAtomicAdd(p, v);   // global_atomic_add_f32
}

// move 8 contiguous f32 -> 8 f16 in LDS (2x global b128, 1x ds_store_b128)
__device__ __forceinline__ void stage8(const float* __restrict__ src, _Float16* dst) {
  float4 f0 = *(const float4*)(src);
  float4 f1 = *(const float4*)(src + 4);
  v8h h;
  h[0] = (_Float16)f0.x; h[1] = (_Float16)f0.y;
  h[2] = (_Float16)f0.z; h[3] = (_Float16)f0.w;
  h[4] = (_Float16)f1.x; h[5] = (_Float16)f1.y;
  h[6] = (_Float16)f1.z; h[7] = (_Float16)f1.w;
  *(v8h*)dst = h;
}

// ---- weight prep: W[k][n] (f32) -> Wt[n][k] (f16), 3 matrices ------------------
__global__ void prep_weights(const float* __restrict__ Wm, const float* __restrict__ We,
                             const float* __restrict__ Wa,
                             _Float16* __restrict__ Tm, _Float16* __restrict__ Te,
                             _Float16* __restrict__ Ta) {
  int i = blockIdx.x * blockDim.x + threadIdx.x;
  const int per = KTOT * Dk;
  if (i >= 3 * per) return;
  int w = i / per, r = i - w * per;
  int k = r / Dk, n = r - k * Dk;
  const float* W = (w == 0) ? Wm : (w == 1) ? We : Wa;
  _Float16* T = (w == 0) ? Tm : (w == 1) ? Te : Ta;
  T[n * KTOT + k] = (_Float16)W[k * Dk + n];
}

__global__ void zero_f32(float* __restrict__ p, long n) {
  long i = (long)blockIdx.x * blockDim.x + threadIdx.x;
  long stride = (long)gridDim.x * blockDim.x;
  for (; i < n; i += stride) p[i] = 0.0f;
}

// ---- stage 1: m = relu([hs|ef|hd] @ W_msg + b), atomic-accumulate mean inputs --
__global__ void __launch_bounds__(BLOCKT) k_msg(
    const int* __restrict__ src_idx, const int* __restrict__ dst_idx,
    const float* __restrict__ src_feats, const float* __restrict__ dst_feats,
    const float* __restrict__ efeats,
    const _Float16* __restrict__ Wt, const float* __restrict__ bias,
    float* __restrict__ hsum, float* __restrict__ deg, int E, int ntiles) {
  __shared__ __align__(16) _Float16 A[16 * APAD];
  const int tid = threadIdx.x;
  const int lane = tid & 31, wave = tid >> 5;
  const int col = (wave << 4) + (lane & 15);
  const int aoff = (lane >> 4) << 3;   // 0 / 8
  const int boff = (lane >> 4) << 3;   // 0 / 8
  const int mbase = (lane >> 4) << 3;  // C rows 0..7 / 8..15
  const _Float16* Arow = A + (lane & 15) * APAD;
  const _Float16* Bp = Wt + col * KTOT + boff;

  for (int t = blockIdx.x; t < ntiles; t += gridDim.x) {
    const int ebase = t << 4;
    // staging: 3 segs x 16 rows x (16 threads x 8 floats) = 768 work items
    for (int idx = tid; idx < 3 * 16 * 16; idx += BLOCKT) {
      int part = (idx & 15) << 3;      // 0..120, step 8
      int rs = idx >> 4;               // 0..47
      int seg = rs >> 4, row = rs & 15;
      int e = ebase + row; if (e >= E) e = E - 1;
      const float* srcp;
      if (seg == 0)      srcp = src_feats + (size_t)src_idx[e] * Dk;
      else if (seg == 1) srcp = efeats + (size_t)e * Dk;
      else               srcp = dst_feats + (size_t)dst_idx[e] * Dk;
      stage8(srcp + part, &A[row * APAD + seg * Dk + part]);
    }
    __syncthreads();

    // full 12-fragment A prefetch: all 24 ds_loads in flight before the WMMA chain
    v8f acc = gemm_tile<NKS>(Arow, Bp, aoff);
    const float bb = bias[col];
#pragma unroll
    for (int r = 0; r < 8; ++r) {
      int e = ebase + mbase + r;
      if (e < E) {
        float v = acc[r] + bb;
        v = v > 0.0f ? v : 0.0f;
        atomAddF(&hsum[(size_t)dst_idx[e] * Dk + col], v);
      }
    }
    if (tid < 16) {
      int e = ebase + tid;
      if (e < E) atomAddF(&deg[dst_idx[e]], 1.0f);
    }
    __syncthreads();
  }
}

// ---- segment mean: out[i] = deg>0 ? sum[i]/deg : 0 -----------------------------
__global__ void k_mean(const float* sum, const float* __restrict__ deg, float* out,
                       long total) {
  long i = (long)blockIdx.x * blockDim.x + threadIdx.x;
  long stride = (long)gridDim.x * blockDim.x;
  for (; i < total; i += stride) {
    float d = deg[i >> 7];
    out[i] = d > 0.0f ? sum[i] / d : 0.0f;
  }
}

// ---- stage 2 fused: eh = relu([hs|ef|hn] @ W_e + b_e); m_out = [hd|hn|eh] @ W_a
__global__ void __launch_bounds__(BLOCKT) k_edge_apply(
    const int* __restrict__ src_idx, const int* __restrict__ dst_idx,
    const float* __restrict__ src_feats, const float* __restrict__ dst_feats,
    const float* __restrict__ efeats, const float* __restrict__ hneigh,
    const _Float16* __restrict__ WtE, const float* __restrict__ bE,
    const _Float16* __restrict__ WtA, const float* __restrict__ bA,
    float* __restrict__ eh_out, float* __restrict__ hosum, int E, int ntiles) {
  __shared__ __align__(16) _Float16 A1[16 * APAD];
  __shared__ __align__(16) _Float16 A2[16 * APAD];
  const int tid = threadIdx.x;
  const int lane = tid & 31, wave = tid >> 5;
  const int col = (wave << 4) + (lane & 15);
  const int aoff = (lane >> 4) << 3;
  const int boff = (lane >> 4) << 3;
  const int mbase = (lane >> 4) << 3;
  const _Float16* Arow1 = A1 + (lane & 15) * APAD;
  const _Float16* Arow2 = A2 + (lane & 15) * APAD;
  const _Float16* BpE = WtE + col * KTOT + boff;
  const _Float16* BpA = WtA + col * KTOT + boff;

  for (int t = blockIdx.x; t < ntiles; t += gridDim.x) {
    const int ebase = t << 4;
    // A1 = [hs | ef | hn_dst] : 768 work items of 8 floats
    for (int idx = tid; idx < 3 * 16 * 16; idx += BLOCKT) {
      int part = (idx & 15) << 3;
      int rs = idx >> 4;
      int seg = rs >> 4, row = rs & 15;
      int e = ebase + row; if (e >= E) e = E - 1;
      const float* srcp;
      if (seg == 0)      srcp = src_feats + (size_t)src_idx[e] * Dk;
      else if (seg == 1) srcp = efeats + (size_t)e * Dk;
      else               srcp = hneigh + (size_t)dst_idx[e] * Dk;
      stage8(srcp + part, &A1[row * APAD + seg * Dk + part]);
    }
    // A2[:, 0:256) = [hd | hn_dst] : 512 work items (eh fills 256:384 after GEMM1)
    for (int idx = tid; idx < 2 * 16 * 16; idx += BLOCKT) {
      int part = (idx & 15) << 3;
      int rs = idx >> 4;
      int seg = rs >> 4, row = rs & 15;
      int e = ebase + row; if (e >= E) e = E - 1;
      const float* srcp = (seg == 0) ? dst_feats + (size_t)dst_idx[e] * Dk
                                     : hneigh + (size_t)dst_idx[e] * Dk;
      stage8(srcp + part, &A2[row * APAD + seg * Dk + part]);
    }
    __syncthreads();

    // GEMM 1: eh tile (group-of-4 A prefetch: bounded pressure, 8 loads/wait)
    v8f acc = gemm_tile<4>(Arow1, BpE, aoff);
    const float bbE = bE[col];
#pragma unroll
    for (int r = 0; r < 8; ++r) {
      int m = mbase + r;
      int e = ebase + m;
      float v = acc[r] + bbE;
      v = v > 0.0f ? v : 0.0f;
      if (e < E) eh_out[(size_t)e * Dk + col] = v;
      A2[m * APAD + 2 * Dk + col] = (_Float16)v;  // distinct (m, col) per lane
    }
    __syncthreads();

    // GEMM 2: m_out tile, accumulate h_out mean inputs
    v8f acc2 = gemm_tile<4>(Arow2, BpA, aoff);
    const float bbA = bA[col];
#pragma unroll
    for (int r = 0; r < 8; ++r) {
      int e = ebase + mbase + r;
      if (e < E) atomAddF(&hosum[(size_t)dst_idx[e] * Dk + col], acc2[r] + bbA);
    }
    __syncthreads();
  }
}

extern "C" void kernel_launch(void* const* d_in, const int* in_sizes, int n_in,
                              void* d_out, int out_size, void* d_ws, size_t ws_size,
                              hipStream_t stream) {
  const int*   src_idx   = (const int*)d_in[0];
  const int*   dst_idx   = (const int*)d_in[1];
  const float* src_feats = (const float*)d_in[2];
  const float* dst_feats = (const float*)d_in[3];
  const float* efeats    = (const float*)d_in[4];
  const float* W_msg     = (const float*)d_in[5];
  const float* b_msg     = (const float*)d_in[6];
  const float* W_edge    = (const float*)d_in[7];
  const float* b_edge    = (const float*)d_in[8];
  const float* W_apply   = (const float*)d_in[9];
  const float* b_apply   = (const float*)d_in[10];

  const int E = in_sizes[0];
  const int N = in_sizes[2] / Dk;

  // workspace carve-out (256B aligned)
  char* ws = (char*)d_ws;
  size_t off = 0;
  auto carve = [&](size_t bytes) {
    void* p = ws + off;
    off = (off + bytes + 255) & ~(size_t)255;
    return p;
  };
  _Float16* Tm     = (_Float16*)carve((size_t)KTOT * Dk * 2);
  _Float16* Te     = (_Float16*)carve((size_t)KTOT * Dk * 2);
  _Float16* Ta     = (_Float16*)carve((size_t)KTOT * Dk * 2);
  float*    hneigh = (float*)carve((size_t)N * Dk * 4);
  float*    hosum  = (float*)carve((size_t)N * Dk * 4);
  float*    deg    = (float*)carve((size_t)N * 4);

  // 1) weights -> f16, transposed [n][k]
  {
    int tot = 3 * KTOT * Dk;
    prep_weights<<<(tot + 255) / 256, 256, 0, stream>>>(W_msg, W_edge, W_apply, Tm, Te, Ta);
  }
  // 2) zero accumulators (must re-zero every call: graph replays reuse d_ws)
  zero_f32<<<1024, 256, 0, stream>>>(hneigh, (long)N * Dk);
  zero_f32<<<1024, 256, 0, stream>>>(hosum, (long)N * Dk);
  zero_f32<<<256, 256, 0, stream>>>(deg, (long)N);

  const int ntiles = (E + 15) / 16;
  const int gTiles = ntiles < 2048 ? ntiles : 2048;

  // 3) message GEMM + mean-accumulate
  k_msg<<<gTiles, BLOCKT, 0, stream>>>(src_idx, dst_idx, src_feats, dst_feats, efeats,
                                       Tm, b_msg, hneigh, deg, E, ntiles);
  // 4) h_neigh = mean (in place)
  {
    long tot = (long)N * Dk;
    int g = (int)((tot + 255) / 256); if (g > 16384) g = 16384;
    k_mean<<<g, 256, 0, stream>>>(hneigh, deg, hneigh, tot);
  }
  // 5) fused edge GEMM (-> eh in d_out) + apply GEMM (-> h_out sums)
  float* eh_out = (float*)d_out + (size_t)N * Dk;
  k_edge_apply<<<gTiles, BLOCKT, 0, stream>>>(src_idx, dst_idx, src_feats, dst_feats,
                                              efeats, hneigh, Te, b_edge, Ta, b_apply,
                                              eh_out, hosum, E, ntiles);
  // 6) h_out = mean -> d_out[0 .. N*Dk)
  {
    long tot = (long)N * Dk;
    int g = (int)((tot + 255) / 256); if (g > 16384) g = 16384;
    k_mean<<<g, 256, 0, stream>>>(hosum, deg, (float*)d_out, tot);
  }
}
